// scaled_dot_product_attn_7215545057961
// MI455X (gfx1250) — matrix operand order repaired and checked by reading, not compiled, not run on hardware
//
#include <hip/hip_runtime.h>

typedef __attribute__((ext_vector_type(16))) __bf16       v16bf;
typedef __attribute__((ext_vector_type(2)))  __bf16       v2bf;
typedef __attribute__((ext_vector_type(8)))  float        v8f;
typedef __attribute__((ext_vector_type(4)))  unsigned int u32x4;
typedef __attribute__((ext_vector_type(2)))  unsigned int u32x2;
typedef __attribute__((ext_vector_type(4)))  float        f32x4;
typedef __attribute__((ext_vector_type(2)))  float        f32x2;

namespace {

constexpr int DK   = 64;    // head dim
constexpr int SEQ  = 2048;  // sequence length
constexpr int KT   = 64;    // keys per tile (4 x 16-key WMMA subtiles)
constexpr int QW   = 32;    // query rows per wave = 2 x 16-query tiles
constexpr int NW   = 8;     // waves per workgroup
constexpr int KSTR = 72;    // Ks row stride (elements): 144B rows, 16B aligned
constexpr int VSTR = 72;    // Vt row stride (elements)

union FragBF {
  u32x4 u[2];
  v16bf v;
};

union PackBF {
  v2bf         b;
  unsigned int u;
};

__device__ inline v16bf ld_frag(const __bf16* p0, const __bf16* p1) {
  FragBF f;
  f.u[0] = *reinterpret_cast<const u32x4*>(p0);
  f.u[1] = *reinterpret_cast<const u32x4*>(p1);
  return f.v;
}

__device__ inline unsigned int pack_bf16(float x, float y) {
  PackBF p;
  p.b = v2bf{(__bf16)x, (__bf16)y};  // -> v_cvt_pk_bf16_f32
  return p.u;
}

// raw v_exp_f32: args are always <= 0 here; flush-to-zero underflow is exactly
// what online softmax wants (no denorm/range fixup code needed).
__device__ inline float fast_exp2(float x) {
#if __has_builtin(__builtin_amdgcn_exp2f)
  return __builtin_amdgcn_exp2f(x);
#else
  float r;
  asm("v_exp_f32 %0, %1" : "=v"(r) : "v"(x));
  return r;
#endif
}

__device__ inline float fast_rcp(float x) {
#if __has_builtin(__builtin_amdgcn_rcpf)
  return __builtin_amdgcn_rcpf(x);
#else
  float r;
  asm("v_rcp_f32 %0, %1" : "=v"(r) : "v"(x));
  return r;
#endif
}

} // namespace

__global__ __launch_bounds__(256) void flash_attn_bf16wmma_kernel(
    const float* __restrict__ Q, const float* __restrict__ K,
    const float* __restrict__ V, float* __restrict__ O) {
  __shared__ __align__(16) __bf16 Ks[KT * KSTR];  // [key][d] bf16
  __shared__ __align__(16) __bf16 Vt[DK * VSTR];  // [d][key] bf16 (transposed)

  const int tid  = threadIdx.x;
  const int wave = tid >> 5;
  const int lane = tid & 31;
  const int half = lane >> 4;
  const int l16  = lane & 15;

  const size_t head = (size_t)blockIdx.y * SEQ * DK;
  const int q0 = blockIdx.x * (NW * QW) + wave * QW;  // 2 query tiles per wave

  // ---- Q^T B-fragments (d x query) for both q-tiles; scale*log2(e) folded ----
  const float qs = 0.125f * 1.44269504088896f;
  FragBF bq[2][2];  // [q-tile g][d-chunk c]
#pragma unroll
  for (int g = 0; g < 2; ++g) {
    const float* qrow = Q + head + (size_t)(q0 + 16 * g + l16) * DK;
#pragma unroll
    for (int c = 0; c < 2; ++c) {
#pragma unroll
      for (int w = 0; w < 4; ++w) {
        const f32x4 f = *reinterpret_cast<const f32x4*>(qrow + 32 * c + 16 * half + 4 * w);
        bq[g][c].v[4 * w + 0] = (__bf16)(f.x * qs);
        bq[g][c].v[4 * w + 1] = (__bf16)(f.y * qs);
        bq[g][c].v[4 * w + 2] = (__bf16)(f.z * qs);
        bq[g][c].v[4 * w + 3] = (__bf16)(f.w * qs);
      }
    }
  }

  v8f o[2][4] = {};                      // O accumulators per q-tile
  float m[2]  = {-1e30f, -1e30f};        // per-lane (query = l16) running max
  float ls[2] = {0.0f, 0.0f};            // per-lane running sum

  for (int kt = 0; kt < SEQ / KT; ++kt) {
    const float* kb = K + head + (size_t)kt * KT * DK;
    const float* vb = V + head + (size_t)kt * KT * DK;

    __syncthreads();  // previous tile fully consumed

    // K tile: coalesced float4 loads -> packed b64 LDS stores (row-major)
#pragma unroll
    for (int i = 0; i < (KT * DK) / (256 * 4); ++i) {
      const int e   = 4 * (tid + 256 * i);  // 4 consecutive d of one key
      const int key = e >> 6;
      const int d   = e & 63;
      const f32x4 f = *reinterpret_cast<const f32x4*>(kb + e);
      const u32x2 kp = {pack_bf16(f.x, f.y), pack_bf16(f.z, f.w)};
      *reinterpret_cast<u32x2*>(&Ks[key * KSTR + d]) = kp;
    }
    // V tile: 2x2 (key,d) blocks -> pack along key -> b32 stores into Vt[d][key]
#pragma unroll
    for (int i = 0; i < (KT * DK) / (256 * 4); ++i) {
      const int b   = tid + 256 * i;
      const int key = (b >> 5) * 2;
      const int d   = (b & 31) * 2;
      const f32x2 g0 = *reinterpret_cast<const f32x2*>(vb + key * DK + d);
      const f32x2 g1 = *reinterpret_cast<const f32x2*>(vb + (key + 1) * DK + d);
      *reinterpret_cast<unsigned int*>(&Vt[(d + 0) * VSTR + key]) = pack_bf16(g0.x, g1.x);
      *reinterpret_cast<unsigned int*>(&Vt[(d + 1) * VSTR + key]) = pack_bf16(g0.y, g1.y);
    }
    if (kt + 1 < SEQ / KT) {  // prefetch next tile
      __builtin_prefetch(kb + KT * DK + tid * 16, 0, 0);
      __builtin_prefetch(vb + KT * DK + tid * 16, 0, 0);
    }
    __syncthreads();

    // ---- S^T = K Q'^T for both q-tiles; each K A-fragment used twice ----
    v8f st[2][4];
#pragma unroll
    for (int t = 0; t < 4; ++t) {
      v8f acc0 = {}, acc1 = {};
#pragma unroll
      for (int c = 0; c < 2; ++c) {
        const int base = (16 * t + l16) * KSTR + 32 * c + 16 * half;
        const v16bf ak = ld_frag(&Ks[base], &Ks[base + 8]);
        acc0 = __builtin_amdgcn_wmma_f32_16x16x32_bf16(false, ak, false, bq[0][c].v,
                                                       (short)0, acc0, false, false);
        acc1 = __builtin_amdgcn_wmma_f32_16x16x32_bf16(false, ak, false, bq[1][c].v,
                                                       (short)0, acc1, false, false);
      }
      st[0][t] = acc0;
      st[1][t] = acc1;
    }

    // ---- softmax update per q-tile (lane-local rows; query = l16) ----
    FragBF ap[2][2];  // [q-tile g][key-chunk j]
#pragma unroll
    for (int g = 0; g < 2; ++g) {
      float mv[8];
#pragma unroll
      for (int v = 0; v < 8; ++v)
        mv[v] = fmaxf(fmaxf(st[g][0][v], st[g][1][v]), fmaxf(st[g][2][v], st[g][3][v]));
      float mx = fmaxf(fmaxf(fmaxf(mv[0], mv[1]), fmaxf(mv[2], mv[3])),
                       fmaxf(fmaxf(mv[4], mv[5]), fmaxf(mv[6], mv[7])));
      mx = fmaxf(mx, __shfl_xor(mx, 16, 32));  // merge the two key-halves
      const float nm    = fmaxf(m[g], mx);
      const float alpha = fast_exp2(m[g] - nm);
      m[g] = nm;

      if (__ballot(alpha != 1.0f)) {  // rescale O only if a row max moved
#pragma unroll
        for (int v = 0; v < 8; ++v) {
          const float av = __shfl(alpha, v + 8 * half, 32);
#pragma unroll
          for (int nc = 0; nc < 4; ++nc) o[g][nc][v] *= av;
        }
      }

      float rs4[4];
#pragma unroll
      for (int t = 0; t < 4; ++t) {
        float r0 = 0.0f, r1 = 0.0f;
#pragma unroll
        for (int v = 0; v < 8; v += 2) {
          const float p0 = fast_exp2(st[g][t][v] - nm);
          const float p1 = fast_exp2(st[g][t][v + 1] - nm);
          r0 += p0;
          r1 += p1;
          ap[g][t >> 1].v[(t & 1) * 8 + v]     = (__bf16)p0;
          ap[g][t >> 1].v[(t & 1) * 8 + v + 1] = (__bf16)p1;
        }
        rs4[t] = r0 + r1;
      }
      float rs = (rs4[0] + rs4[1]) + (rs4[2] + rs4[3]);
      rs += __shfl_xor(rs, 16, 32);
      ls[g] = ls[g] * alpha + rs;
    }

    // ---- O += P V : each V B-fragment used twice ----
#pragma unroll
    for (int nc = 0; nc < 4; ++nc) {
#pragma unroll
      for (int j = 0; j < 2; ++j) {
        const int base = (16 * nc + l16) * VSTR + 32 * j + 8 * half;
        const v16bf bv = ld_frag(&Vt[base], &Vt[base + 16]);
        o[0][nc] = __builtin_amdgcn_wmma_f32_16x16x32_bf16(false, ap[0][j].v, false, bv,
                                                           (short)0, o[0][nc], false, false);
        o[1][nc] = __builtin_amdgcn_wmma_f32_16x16x32_bf16(false, ap[1][j].v, false, bv,
                                                           (short)0, o[1][nc], false, false);
      }
    }
  }

  // ---- normalize (broadcast row sums into O layout) and store fp32 ----
#pragma unroll
  for (int g = 0; g < 2; ++g) {
#pragma unroll
    for (int v = 0; v < 8; ++v) {
      const float lv  = __shfl(ls[g], v + 8 * half, 32);
      const float inv = fast_rcp(lv);
      float* orow = O + head + (size_t)(q0 + 16 * g + v + 8 * half) * DK;
#pragma unroll
      for (int nc = 0; nc < 4; ++nc)
        orow[16 * nc + l16] = o[g][nc][v] * inv;
    }
  }
}

extern "C" void kernel_launch(void* const* d_in, const int* /*in_sizes*/, int /*n_in*/,
                              void* d_out, int /*out_size*/, void* /*d_ws*/,
                              size_t /*ws_size*/, hipStream_t stream) {
  const float* Q = (const float*)d_in[0];
  const float* K = (const float*)d_in[1];
  const float* V = (const float*)d_in[2];
  float* O = (float*)d_out;

  dim3 grid(SEQ / (NW * QW), 4 * 16, 1);  // 8 q-blocks x (B*H)=64 heads
  dim3 block(256, 1, 1);                  // 8 waves of 32
  flash_attn_bf16wmma_kernel<<<grid, block, 0, stream>>>(Q, K, V, O);
}
